// BitLinear_26405458936545
// MI455X (gfx1250) — compile-verified, matrix operand order checked by reading, not run on hardware
//
#include <hip/hip_runtime.h>
#include <stdint.h>

typedef __attribute__((ext_vector_type(8))) int v8i;

#define M_TOK 8192
#define K_DIM 4096
#define N_OUT 16384
#define EPSQ 1e-5f
#define NW_ELEMS 67108864.0f   // 16384*4096

// ---------------- scratch zero ----------------
__global__ __launch_bounds__(256) void zero_ws(float* p) {
  p[threadIdx.x] = 0.0f;
}

// ---------------- per-tensor |w| sum ----------------
__global__ __launch_bounds__(256) void w_abssum(const float4* __restrict__ w4,
                                                float* __restrict__ absSum, int n4) {
  float s = 0.0f;
  for (int i = blockIdx.x * blockDim.x + threadIdx.x; i < n4;
       i += gridDim.x * blockDim.x) {
    float4 v = w4[i];
    s += fabsf(v.x) + fabsf(v.y) + fabsf(v.z) + fabsf(v.w);
  }
  __shared__ float red[256];
  red[threadIdx.x] = s;
  __syncthreads();
  for (int off = 128; off > 0; off >>= 1) {
    if ((int)threadIdx.x < off) red[threadIdx.x] += red[threadIdx.x + off];
    __syncthreads();
  }
  if (threadIdx.x == 0) atomicAdd(absSum, red[0]);
}

__device__ inline signed char quant1(float v, float r, float lo, float hi) {
  float q = rintf(v * r);
  q = fminf(fmaxf(q, lo), hi);
  return (signed char)(int)q;
}

// ---------------- ternary weight quant ----------------
__global__ __launch_bounds__(256) void w_quant(const float4* __restrict__ w4,
                                               char4* __restrict__ wq4,
                                               const float* __restrict__ absSum,
                                               int n4) {
  int i = blockIdx.x * blockDim.x + threadIdx.x;
  if (i >= n4) return;
  float scale = fmaxf(absSum[0] * (1.0f / NW_ELEMS), EPSQ);
  float r = 1.0f / scale;
  float4 v = w4[i];
  char4 q;
  q.x = quant1(v.x, r, -1.f, 1.f);
  q.y = quant1(v.y, r, -1.f, 1.f);
  q.z = quant1(v.z, r, -1.f, 1.f);
  q.w = quant1(v.w, r, -1.f, 1.f);
  wq4[i] = q;
}

// ---------------- per-token activation quant ----------------
__global__ __launch_bounds__(256) void act_quant(const float* __restrict__ x,
                                                 signed char* __restrict__ xq,
                                                 float* __restrict__ tokScale,
                                                 const float* __restrict__ absSum) {
  const int tok = blockIdx.x;
  const float4* row = (const float4*)(x + (size_t)tok * K_DIM);
  char4* qrow = (char4*)(xq + (size_t)tok * K_DIM);
  float4 v[4];
  float m = 0.0f;
  for (int c = 0; c < 4; ++c) {
    v[c] = row[threadIdx.x * 4 + c];
    m = fmaxf(m, fmaxf(fmaxf(fabsf(v[c].x), fabsf(v[c].y)),
                       fmaxf(fabsf(v[c].z), fabsf(v[c].w))));
  }
  __shared__ float red[256];
  red[threadIdx.x] = m;
  __syncthreads();
  for (int off = 128; off > 0; off >>= 1) {
    if ((int)threadIdx.x < off)
      red[threadIdx.x] = fmaxf(red[threadIdx.x], red[threadIdx.x + off]);
    __syncthreads();
  }
  float amax = fmaxf(red[0], EPSQ);
  float s = 128.0f / amax;
  for (int c = 0; c < 4; ++c) {
    char4 q;
    q.x = quant1(v[c].x, s, -128.f, 127.f);
    q.y = quant1(v[c].y, s, -128.f, 127.f);
    q.z = quant1(v[c].z, s, -128.f, 127.f);
    q.w = quant1(v[c].w, s, -128.f, 127.f);
    qrow[threadIdx.x * 4 + c] = q;
  }
  if (threadIdx.x == 0) {
    float wscale = fmaxf(absSum[0] * (1.0f / NW_ELEMS), EPSQ);
    tokScale[tok] = wscale * amax * (1.0f / 128.0f);
  }
}

// ---------------- int8 WMMA GEMM ----------------
__device__ inline void async_cp16(unsigned lds, const signed char* g) {
  asm volatile("global_load_async_to_lds_b128 %0, %1, off"
               :: "v"(lds), "v"((unsigned long long)(uintptr_t)g)
               : "memory");
}

__device__ inline void wait_async0() {
#if __has_builtin(__builtin_amdgcn_s_wait_asynccnt)
  __builtin_amdgcn_s_wait_asynccnt(0);
#else
  asm volatile("s_wait_asynccnt 0" ::: "memory");
#endif
}

// Gather one 16x64 int8 fragment (A layout; B uses the same per-lane K-packing
// with n playing the role of m) from an LDS tile stored as [row][64B of K].
// Lane (lr,hi): VGPR pairs live at byte offsets hi*8 + {0,16,32,48}.
__device__ inline v8i loadFrag(const signed char* rowBase, int hi) {
  const uint2* p = (const uint2*)(rowBase + hi * 8);
  uint2 q0 = p[0], q1 = p[2], q2 = p[4], q3 = p[6];
  v8i f;
  f[0] = (int)q0.x; f[1] = (int)q0.y;
  f[2] = (int)q1.x; f[3] = (int)q1.y;
  f[4] = (int)q2.x; f[5] = (int)q2.y;
  f[6] = (int)q3.x; f[7] = (int)q3.y;
  return f;
}

// Block tile 256(M) x 128(N), K-step 64, 8 waves as 4(M) x 2(N),
// wave tile 64x64 = 4x4 WMMA accumulators (16 wmma per K-step per wave).
__global__ __launch_bounds__(256) void bit_gemm(const signed char* __restrict__ Aq,
                                                const signed char* __restrict__ Wq,
                                                const float* __restrict__ tokScale,
                                                float* __restrict__ out) {
  __shared__ signed char As[2 * 256 * 64];  // [buf][m:256][k:64]  2x16KB
  __shared__ signed char Bs[2 * 128 * 64];  // [buf][n:128][k:64]  2x8KB

  const int tid = threadIdx.x;
  const int mBase = blockIdx.y * 256;
  const int nBase = blockIdx.x * 128;

  const unsigned asBase = (unsigned)(uintptr_t)(void*)As;
  const unsigned bsBase = (unsigned)(uintptr_t)(void*)Bs;

  // Tile loaders: A = 16KB -> 4 chunks of 16B/thread; B = 8KB -> 2 chunks.
  const int r0 = tid >> 2;            // rows 0..63
  const int c0 = (tid & 3) * 16;      // 16B column within 64B K-row
  const unsigned ldsR0 = (unsigned)(r0 * 64 + c0);
  const signed char* gA0 = Aq + (size_t)(mBase + r0) * K_DIM + c0;
  const signed char* gB0 = Wq + (size_t)(nBase + r0) * K_DIM + c0;

  // Wave decomposition.
  const int lane = tid & 31;
  const int wv = tid >> 5;
  const int waveM = wv & 3;   // 4 waves over M: 64 rows each
  const int waveN = wv >> 2;  // 2 waves over N: 64 cols each
  const int lr = lane & 15;
  const int hi = lane >> 4;

  const v8i vzero = {0, 0, 0, 0, 0, 0, 0, 0};
  v8i acc[4][4];
  for (int mi = 0; mi < 4; ++mi)
    for (int ni = 0; ni < 4; ++ni) acc[mi][ni] = vzero;

  auto issue = [&](int buf, int k0) {
    unsigned aoff = asBase + (unsigned)buf * 16384u;
    unsigned boff = bsBase + (unsigned)buf * 8192u;
#pragma unroll
    for (int j = 0; j < 4; ++j)
      async_cp16(aoff + ldsR0 + (unsigned)j * 64u * 64u,
                 gA0 + (size_t)j * 64 * K_DIM + k0);
#pragma unroll
    for (int j = 0; j < 2; ++j)
      async_cp16(boff + ldsR0 + (unsigned)j * 64u * 64u,
                 gB0 + (size_t)j * 64 * K_DIM + k0);
  };

  issue(0, 0);
  wait_async0();
  __syncthreads();

  const int KT = K_DIM / 64;  // 64 K-steps
  for (int kt = 0; kt < KT; ++kt) {
    const int buf = kt & 1;
    if (kt + 1 < KT) issue(buf ^ 1, (kt + 1) * 64);

    const signed char* AsB = As + buf * 16384;
    const signed char* BsB = Bs + buf * 8192;

    v8i a[4], b[4];
#pragma unroll
    for (int mi = 0; mi < 4; ++mi)
      a[mi] = loadFrag(AsB + (waveM * 64 + mi * 16 + lr) * 64, hi);
#pragma unroll
    for (int ni = 0; ni < 4; ++ni)
      b[ni] = loadFrag(BsB + (waveN * 64 + ni * 16 + lr) * 64, hi);

#pragma unroll
    for (int mi = 0; mi < 4; ++mi)
#pragma unroll
      for (int ni = 0; ni < 4; ++ni)
        acc[mi][ni] = __builtin_amdgcn_wmma_i32_16x16x64_iu8(
            true, a[mi], true, b[ni], acc[mi][ni], false, false);

    wait_async0();
    __syncthreads();
  }

  // Epilogue: i32 -> f32 with per-token combined scale.
  // C/D layout: VGPR r holds M=r (lanes 0-15) / M=r+8 (lanes 16-31), N=lr.
  for (int mi = 0; mi < 4; ++mi) {
    const int mrow0 = mBase + waveM * 64 + mi * 16 + hi * 8;
    float sc[8];
#pragma unroll
    for (int r = 0; r < 8; ++r) sc[r] = tokScale[mrow0 + r];
    for (int ni = 0; ni < 4; ++ni) {
      const int n = nBase + waveN * 64 + ni * 16 + lr;
#pragma unroll
      for (int r = 0; r < 8; ++r) {
        out[(size_t)(mrow0 + r) * N_OUT + n] = (float)acc[mi][ni][r] * sc[r];
      }
    }
  }
}

// ---------------- host launcher ----------------
extern "C" void kernel_launch(void* const* d_in, const int* in_sizes, int n_in,
                              void* d_out, int out_size, void* d_ws, size_t ws_size,
                              hipStream_t stream) {
  const float* x = (const float*)d_in[0];   // [4,2048,4096]
  const float* w = (const float*)d_in[1];   // [16384,4096]
  float* out = (float*)d_out;               // [4,2048,16384]
  char* ws = (char*)d_ws;

  float* absSum = (float*)ws;                         // 1 KB zeroed region
  float* tokScale = (float*)(ws + 1024);              // 8192 floats
  signed char* xq = (signed char*)(ws + (1 << 16));   // 32 MB
  signed char* wq = xq + (size_t)M_TOK * K_DIM;       // 64 MB

  const int n4w = (N_OUT / 4) * K_DIM;  // 16,777,216 float4s

  zero_ws<<<1, 256, 0, stream>>>(absSum);
  w_abssum<<<4096, 256, 0, stream>>>((const float4*)w, absSum, n4w);
  w_quant<<<n4w / 256, 256, 0, stream>>>((const float4*)w, (char4*)wq, absSum, n4w);
  act_quant<<<M_TOK, 256, 0, stream>>>(x, xq, tokScale, absSum);

  dim3 grid(N_OUT / 128, M_TOK / 256);  // (128, 32)
  bit_gemm<<<grid, 256, 0, stream>>>(xq, wq, tokScale, out);
}
